// AttentionHead_50594714747521
// MI455X (gfx1250) — compile-verified, matrix operand order checked by reading, not run on hardware
//
#include <hip/hip_runtime.h>

#define D_MODEL 1024
#define D_HEAD  64
#define BATCH   4
#define SEQ     2048

typedef __attribute__((ext_vector_type(16))) _Float16 v16h;
typedef __attribute__((ext_vector_type(8)))  float    v8f;

union HV {
    v16h  v;
    uint4 q[2];
    _Float16 h[16];
};

__device__ __forceinline__ v8f wmma_f16(const HV& a, const HV& b, v8f c) {
    // D = A(16x32 f16) * B(32x16 f16) + C(16x16 f32)
    return __builtin_amdgcn_wmma_f32_16x16x32_f16(false, a.v, false, b.v,
                                                  (short)0, c, false, false);
}

__device__ __forceinline__ float fast_exp2(float x) {
#if __has_builtin(__builtin_amdgcn_exp2f)
    return __builtin_amdgcn_exp2f(x);     // single v_exp_f32
#else
    return exp2f(x);
#endif
}

// LDS byte address of a __shared__ object (flat shared ptr low 32 bits = LDS offset)
__device__ __forceinline__ uint32_t lds_addr_of(const void* p) {
    return (uint32_t)(uintptr_t)p;
}

// CDNA5 async memory->LDS copy, 16B per lane, tracked by ASYNCcnt.
__device__ __forceinline__ void async_copy_b128(uint32_t lds_byte, uint64_t gbase,
                                                uint32_t gbyte_off) {
    asm volatile("global_load_async_to_lds_b128 %0, %1, %2 offset:0"
                 :: "v"(lds_byte), "v"(gbyte_off), "s"(gbase)
                 : "memory");
}

// ---------------------------------------------------------------------------
// Kernel 1: fused QKV projection.
//   E (B*S x 1024 f32) x {Wq,Wk,Wv} (1024 x 64 f32) + bias -> f16 workspace.
//   Q, K stored row-major [b][s][64]; V stored TRANSPOSED [b][d][s] so the
//   attention kernel's PV B-fragments are contiguous along the key dim.
// Block = 256 threads (8 waves), computes 128 rows x (64*3) cols.
// ---------------------------------------------------------------------------
__global__ __launch_bounds__(256, 1)
void qkv_proj_kernel(const float* __restrict__ E,
                     const float* __restrict__ Wq, const float* __restrict__ bq,
                     const float* __restrict__ Wk, const float* __restrict__ bk,
                     const float* __restrict__ Wv, const float* __restrict__ bv,
                     _Float16* __restrict__ Qh, _Float16* __restrict__ Kh,
                     _Float16* __restrict__ Vt)
{
    __shared__ _Float16 eT[128][40];   // E tile, row-major (row, k), pad->80B rows
    __shared__ _Float16 wT[192][40];   // W chunk, TRANSPOSED (n, k)

    const int tid  = threadIdx.x;
    const int w    = tid >> 5;
    const int lane = tid & 31;
    const int n0   = lane & 15;
    const int hi   = lane >> 4;
    const int r0   = blockIdx.x * 128;        // global row of this block
    const int bidx = r0 / SEQ;                // batch (blocks never straddle)

    v8f acc[12] = {};                          // wave: 16 rows x 12 col-tiles

    for (int kc = 0; kc < D_MODEL; kc += 32) {
        // ---- stage E tile (128 x 32 f32 -> f16) ----
        {
            const int rg = tid >> 3;
            const int cg = (tid & 7) * 4;
#pragma unroll
            for (int i = 0; i < 4; ++i) {
                const int r = rg + i * 32;
                const float4 e = *(const float4*)(E + (size_t)(r0 + r) * D_MODEL + kc + cg);
                union { _Float16 h[4]; uint2 u; } pk;
                pk.h[0] = (_Float16)e.x; pk.h[1] = (_Float16)e.y;
                pk.h[2] = (_Float16)e.z; pk.h[3] = (_Float16)e.w;
                *(uint2*)&eT[r][cg] = pk.u;
            }
        }
        // ---- stage W chunk (32 x 192 f32 -> f16, transposed to (n,k)) ----
        {
#pragma unroll
            for (int i = 0; i < 6; ++i) {
                const int idx = tid + i * 256;       // 0..1535 float4 groups
                const int mat = idx >> 9;            // 0=Q 1=K 2=V
                const int rem = idx & 511;
                const int k   = rem >> 4;            // 0..31
                const int n4  = (rem & 15) * 4;      // 0..60
                const float* W = (mat == 0) ? Wq : ((mat == 1) ? Wk : Wv);
                const float4 wv4 = *(const float4*)(W + (size_t)(kc + k) * D_HEAD + n4);
                wT[mat * 64 + n4 + 0][k] = (_Float16)wv4.x;
                wT[mat * 64 + n4 + 1][k] = (_Float16)wv4.y;
                wT[mat * 64 + n4 + 2][k] = (_Float16)wv4.z;
                wT[mat * 64 + n4 + 3][k] = (_Float16)wv4.w;
            }
        }
        __syncthreads();

        // ---- A fragment: wave's 16 rows x 32 k (f16 16x32 layout) ----
        HV a;
        a.q[0] = *(const uint4*)&eT[16 * w + n0][hi * 8];
        a.q[1] = *(const uint4*)&eT[16 * w + n0][hi * 8 + 16];
#pragma unroll
        for (int c = 0; c < 12; ++c) {
            HV bf;   // B fragment: column = output col, contiguous k in wT
            bf.q[0] = *(const uint4*)&wT[16 * c + n0][hi * 8];
            bf.q[1] = *(const uint4*)&wT[16 * c + n0][hi * 8 + 16];
            acc[c] = wmma_f16(a, bf, acc[c]);
        }
        __syncthreads();
    }

    // ---- epilogue: bias + store f16 to workspace ----
#pragma unroll
    for (int c = 0; c < 12; ++c) {
        const int mat = c >> 2;
        const int nl  = (c & 3) * 16 + n0;
        const float* bias = (mat == 0) ? bq : ((mat == 1) ? bk : bv);
        const float bvv = bias[nl];
#pragma unroll
        for (int j = 0; j < 8; ++j) {
            const int m    = j + 8 * hi;               // C layout: lane row
            const int grow = r0 + 16 * w + m;          // global row [0, B*S)
            const _Float16 hval = (_Float16)(acc[c][j] + bvv);
            if (mat == 0) {
                Qh[(size_t)grow * D_HEAD + nl] = hval;
            } else if (mat == 1) {
                Kh[(size_t)grow * D_HEAD + nl] = hval;
            } else {
                const int srow = grow & (SEQ - 1);
                Vt[((size_t)bidx * D_HEAD + nl) * SEQ + srow] = hval;
            }
        }
    }
}

// ---------------------------------------------------------------------------
// Kernel 2: causal flash attention.
//   Grid = (S/128 q-blocks, B). 256 threads; wave w owns 16 query rows.
//   64-key blocks double-buffered in LDS via GLOBAL_LOAD_ASYNC_TO_LDS_B128
//   (ASYNCcnt); online softmax in the exp2 domain with half-wave shuffles;
//   P converted C-layout -> A-layout through a per-wave LDS slab.
// ---------------------------------------------------------------------------
__global__ __launch_bounds__(256, 1)
void flash_attn_kernel(const _Float16* __restrict__ Qh,
                       const _Float16* __restrict__ Kh,
                       const _Float16* __restrict__ Vt,
                       float* __restrict__ out)
{
    __shared__ _Float16 Kt[2][64][72];    // (key, d), double buffered
    __shared__ _Float16 Vl[2][64][72];    // (d, key), double buffered
    __shared__ _Float16 Pl[8][16][72];    // per-wave P staging (row, key)

    const int tid  = threadIdx.x;
    const int w    = tid >> 5;
    const int lane = tid & 31;
    const int n0   = lane & 15;
    const int hi   = lane >> 4;
    const int b    = blockIdx.y;
    const int qblk = blockIdx.x;
    const int qbase = qblk * 128;

    // 1/sqrt(64) * log2(e): score scale folded into exp2-domain softmax
    const float SCALE_LOG2 = 0.125f * 1.44269504088896f;

    // ---- Q A-fragments, resident for the whole kernel ----
    HV qf[2];
    {
        const _Float16* qp = Qh + ((size_t)b * SEQ + qbase + 16 * w + n0) * D_HEAD;
#pragma unroll
        for (int c = 0; c < 2; ++c) {
            qf[c].q[0] = *(const uint4*)(qp + c * 32 + hi * 8);
            qf[c].q[1] = *(const uint4*)(qp + c * 32 + hi * 8 + 16);
        }
    }

    v8f Oacc[4] = {};
    float mrow[8], lrow[8];
#pragma unroll
    for (int j = 0; j < 8; ++j) { mrow[j] = -1e30f; lrow[j] = 0.0f; }

    const uint64_t kbase_g = (uint64_t)(uintptr_t)(Kh + (size_t)b * SEQ * D_HEAD);
    const uint64_t vbase_g = (uint64_t)(uintptr_t)(Vt + (size_t)b * D_HEAD * SEQ);
    const int sr  = tid >> 2;       // staging row 0..63
    const int sc0 = tid & 3;

    // issue one key-block's async copies (4 x b128 per thread -> 16KB total)
    auto stage_async = [&](int blk) {
        const int bufb = blk & 1;
        const int kb2  = blk * 64;
#pragma unroll
        for (int i = 0; i < 2; ++i) {
            const int ck = (sc0 + 4 * i) * 8;                 // halves
            async_copy_b128(lds_addr_of(&Kt[bufb][sr][ck]), kbase_g,
                            (uint32_t)(((kb2 + sr) * D_HEAD + ck) * 2));
            async_copy_b128(lds_addr_of(&Vl[bufb][sr][ck]), vbase_g,
                            (uint32_t)((sr * SEQ + kb2 + ck) * 2));
        }
    };

    const int nkb = 2 * qblk + 2;          // covers keys <= qbase+127, uniform
    stage_async(0);                        // prologue
    for (int kb = 0; kb < nkb; ++kb) {
        const int cur = kb & 1;
        if (kb + 1 < nkb) {
            stage_async(kb + 1);           // overlap next block with compute
            // async loads finish in order: <=4 outstanding => current landed
            asm volatile("s_wait_asynccnt 0x4" ::: "memory");
        } else {
            asm volatile("s_wait_asynccnt 0x0" ::: "memory");
        }
        __syncthreads();

        const int kbase = kb * 64;

        // ---- scores (exp2 domain): S = (Q K^T) * log2(e)/sqrt(d), causal ----
        float sc[4][8];
#pragma unroll
        for (int t = 0; t < 4; ++t) {
            HV k0, k1;                     // B frags: col = key, k = d
            const int kr = 16 * t + n0;
            k0.q[0] = *(const uint4*)&Kt[cur][kr][hi * 8];
            k0.q[1] = *(const uint4*)&Kt[cur][kr][hi * 8 + 16];
            k1.q[0] = *(const uint4*)&Kt[cur][kr][32 + hi * 8];
            k1.q[1] = *(const uint4*)&Kt[cur][kr][32 + hi * 8 + 16];
            v8f s8 = {};
            s8 = wmma_f16(qf[0], k0, s8);
            s8 = wmma_f16(qf[1], k1, s8);
            const int kidx = kbase + 16 * t + n0;
#pragma unroll
            for (int j = 0; j < 8; ++j) {
                const int qr = qbase + 16 * w + j + 8 * hi;
                sc[t][j] = (kidx <= qr) ? s8[j] * SCALE_LOG2 : -1e30f;
            }
        }

        // ---- online softmax: row stats via half-wave xor-shuffles ----
        float al[8];
#pragma unroll
        for (int j = 0; j < 8; ++j) {
            float mx = sc[0][j];
#pragma unroll
            for (int t = 1; t < 4; ++t) mx = fmaxf(mx, sc[t][j]);
#pragma unroll
            for (int d = 1; d < 16; d <<= 1) mx = fmaxf(mx, __shfl_xor(mx, d, 32));
            const float mn = fmaxf(mrow[j], mx);
            al[j] = fast_exp2(mrow[j] - mn);
            float s = 0.0f;
#pragma unroll
            for (int t = 0; t < 4; ++t) {
                const float p = fast_exp2(sc[t][j] - mn);
                sc[t][j] = p;
                s += p;
            }
#pragma unroll
            for (int d = 1; d < 16; d <<= 1) s += __shfl_xor(s, d, 32);
            lrow[j] = lrow[j] * al[j] + s;
            mrow[j] = mn;
        }
#pragma unroll
        for (int dt = 0; dt < 4; ++dt)
#pragma unroll
            for (int j = 0; j < 8; ++j) Oacc[dt][j] *= al[j];

        // ---- P: C-layout -> A-layout via per-wave LDS slab ----
#pragma unroll
        for (int t = 0; t < 4; ++t)
#pragma unroll
            for (int j = 0; j < 8; ++j)
                Pl[w][j + 8 * hi][16 * t + n0] = (_Float16)sc[t][j];
        asm volatile("s_wait_dscnt 0x0" ::: "memory");   // P visible wave-wide

        // ---- O += P V ----
#pragma unroll
        for (int c4 = 0; c4 < 2; ++c4) {
            HV pf;                          // A frag: row = q row, k = key
            pf.q[0] = *(const uint4*)&Pl[w][n0][c4 * 32 + hi * 8];
            pf.q[1] = *(const uint4*)&Pl[w][n0][c4 * 32 + hi * 8 + 16];
#pragma unroll
            for (int dt = 0; dt < 4; ++dt) {
                HV vf;                      // B frag: col = d, k = key (contig)
                vf.q[0] = *(const uint4*)&Vl[cur][dt * 16 + n0][c4 * 32 + hi * 8];
                vf.q[1] = *(const uint4*)&Vl[cur][dt * 16 + n0][c4 * 32 + hi * 8 + 16];
                Oacc[dt] = wmma_f16(pf, vf, Oacc[dt]);
            }
        }
        __syncthreads();   // done with buf[cur] before it is restaged
    }

    // ---- normalize and store f32 output ----
#pragma unroll
    for (int dt = 0; dt < 4; ++dt)
#pragma unroll
        for (int j = 0; j < 8; ++j) {
            const int qr = qbase + 16 * w + j + 8 * hi;
            out[((size_t)b * SEQ + qr) * D_HEAD + dt * 16 + n0] = Oacc[dt][j] / lrow[j];
        }
}

// ---------------------------------------------------------------------------
extern "C" void kernel_launch(void* const* d_in, const int* in_sizes, int n_in,
                              void* d_out, int out_size, void* d_ws, size_t ws_size,
                              hipStream_t stream)
{
    const float* E  = (const float*)d_in[0];
    const float* Wq = (const float*)d_in[1];
    const float* bq = (const float*)d_in[2];
    const float* Wk = (const float*)d_in[3];
    const float* bk = (const float*)d_in[4];
    const float* Wv = (const float*)d_in[5];
    const float* bv = (const float*)d_in[6];

    _Float16* wsQ = (_Float16*)d_ws;                       // [B][S][64] f16
    _Float16* wsK = wsQ + (size_t)BATCH * SEQ * D_HEAD;    // [B][S][64] f16
    _Float16* wsV = wsK + (size_t)BATCH * SEQ * D_HEAD;    // [B][64][S] f16 (transposed)

    qkv_proj_kernel<<<dim3((BATCH * SEQ) / 128), dim3(256), 0, stream>>>(
        E, Wq, bq, Wk, bk, Wv, bv, wsQ, wsK, wsV);

    flash_attn_kernel<<<dim3(SEQ / 128, BATCH), dim3(256), 0, stream>>>(
        wsQ, wsK, wsV, (float*)d_out);
}
